// SudokuRRN_64330020159585
// MI455X (gfx1250) — compile-verified, matrix operand order checked by reading, not run on hardware
//
#include <hip/hip_runtime.h>

// ---------------- problem constants ----------------
#define BB 1024
#define NN 81
#define EE 1620
#define HH 128
#define MM 128
#define NSTEPS 8
#define RR (BB * NN)        // 82944 flat rows
#define G3 (3 * HH)         // 384

typedef __attribute__((ext_vector_type(16))) __bf16 v16bf;
typedef __attribute__((ext_vector_type(8)))  __bf16 v8bf;
typedef __attribute__((ext_vector_type(8)))  float  v8f;

static __device__ __forceinline__ v8f zerov8() {
    v8f z = {0.f, 0.f, 0.f, 0.f, 0.f, 0.f, 0.f, 0.f};
    return z;
}

// Load one 16x32 bf16 operand fragment (A layout == B layout for D = A * W^T)
// from a row-major matrix: lane[3:0] = row, lane[4] selects K-half.
// Per ISA 7.12.2: lanes 0-15 hold K = k0+0..7 and k0+16..23,
//                 lanes 16-31 hold K = k0+8..15 and k0+24..31.
static __device__ __forceinline__ v16bf load_frag(const __bf16* base, int stride, int k0) {
    const int lane = threadIdx.x & 31;
    const int row  = lane & 15;
    const int kb   = (lane >> 4) << 3;
    const __bf16* p = base + (size_t)row * (size_t)stride + (k0 + kb);
    v8bf lo = *(const v8bf*)(p);
    v8bf hi = *(const v8bf*)(p + 16);
    return __builtin_shufflevector(lo, hi, 0, 1, 2, 3, 4, 5, 6, 7,
                                           8, 9, 10, 11, 12, 13, 14, 15);
}

static __device__ __forceinline__ v8f wmma_bf16(v16bf a, v16bf b, v8f c) {
    return __builtin_amdgcn_wmma_f32_16x16x32_bf16(false, a, false, b,
                                                   (short)0, c, false, false);
}

// ---------------- CDNA5 async Global->LDS path (ASYNCcnt) ----------------
// Copies a 16-row x 128-bf16 tile (256B per row) into LDS using
// GLOBAL_LOAD_ASYNC_TO_LDS_B128: 256 x 16B units, 128 threads x 2 issues.
static __device__ __forceinline__ void async_copy_tile_b128(
        void* lds_dst, const __bf16* gsrc, int rowstride_elems) {
    const int t = threadIdx.x;
#pragma unroll
    for (int u = t; u < 256; u += 128) {
        const int row = u >> 4;
        const int seg = u & 15;
        unsigned int loff = (unsigned int)(size_t)lds_dst
                          + (unsigned int)(row * 256 + seg * 16);
        const void* g = (const char*)gsrc + (size_t)row * rowstride_elems * 2 + seg * 16;
        asm volatile("global_load_async_to_lds_b128 %0, %1, off"
                     :: "v"(loff), "v"(g) : "memory");
    }
}

static __device__ __forceinline__ void wait_async0() {
    asm volatile("s_wait_asynccnt 0x0" ::: "memory");
}

// ---------------- CSR build (single thread, deterministic) ----------------
__global__ void build_csr_kernel(const int* __restrict__ edge_index,
                                 int* __restrict__ csr_ptr,
                                 int* __restrict__ csr_edge) {
    if (threadIdx.x != 0 || blockIdx.x != 0) return;
    int cnt[NN];
    for (int i = 0; i < NN; ++i) cnt[i] = 0;
    for (int e = 0; e < EE; ++e) cnt[edge_index[EE + e]]++;
    csr_ptr[0] = 0;
    for (int i = 0; i < NN; ++i) csr_ptr[i + 1] = csr_ptr[i] + cnt[i];
    int pos[NN];
    for (int i = 0; i < NN; ++i) pos[i] = csr_ptr[i];
    for (int e = 0; e < EE; ++e) {
        int r = edge_index[EE + e];
        csr_edge[pos[r]++] = e;
    }
}

// ---------------- f32 -> bf16 weight conversion ----------------
__global__ void cvt_bf16_kernel(const float* __restrict__ src,
                                __bf16* __restrict__ dst, int n) {
    int i = blockIdx.x * blockDim.x + threadIdx.x;
    if (i < n) dst[i] = (__bf16)src[i];
}

// ---------------- input projection: h = x @ W_in^T + b_in ----------------
__global__ __launch_bounds__(256) void inproj_kernel(
        const float* __restrict__ x, const float* __restrict__ Win,
        const float* __restrict__ bin,
        float* __restrict__ hf, __bf16* __restrict__ hbf) {
    size_t gid = (size_t)blockIdx.x * 256 + threadIdx.x;
    if (gid >= (size_t)RR * HH) return;
    int row = (int)(gid >> 7);
    int c   = (int)(gid & 127);
    float s = bin[c];
#pragma unroll
    for (int k = 0; k < 10; ++k)
        s += x[(size_t)row * 10 + k] * Win[c * 10 + k];
    hf[gid]  = s;
    hbf[gid] = (__bf16)s;
}

// ---------------- fused message MLP + deterministic aggregation ----------------
// block = (node n, 16-batch tile); 4 waves x 32 output columns.
// Sender tiles are double-buffered in LDS via async Global->LDS DMA,
// prefetched one edge ahead of the WMMA pipeline.
__global__ __launch_bounds__(128) void msg_agg_kernel(
        const __bf16* __restrict__ hbf,
        const __bf16* __restrict__ Wm1b, const __bf16* __restrict__ Wm2b,
        const float* __restrict__ bm1, const float* __restrict__ bm2,
        const int* __restrict__ edge_index,
        const int* __restrict__ csr_ptr, const int* __restrict__ csr_edge,
        __bf16* __restrict__ aggb) {
    const int n    = blockIdx.x;
    const int b0   = blockIdx.y * 16;
    const int wave = threadIdx.x >> 5;
    const int lane = threadIdx.x & 31;
    const int j0   = wave * 32;

    __shared__ __bf16 sS[2][16][HH];  // double-buffered sender tile (8 KB)
    __shared__ __bf16 sR[16][HH];     // receiver tile (4 KB, block-invariant)
    __shared__ __bf16 sT[16][MM];     // GEMM1 -> GEMM2 staging (4 KB)

    v8f acc[2] = {zerov8(), zerov8()};

    const int e_beg = csr_ptr[n];
    const int e_end = csr_ptr[n + 1];
    const int rowstride = NN * HH;    // flat-row stride over batch index

    // stage the block-invariant receiver tile + first sender tile (async DMA)
    async_copy_tile_b128(&sR[0][0], hbf + ((size_t)b0 * NN + n) * HH, rowstride);
    if (e_beg < e_end) {
        const int s0 = edge_index[csr_edge[e_beg]];
        async_copy_tile_b128(&sS[0][0][0], hbf + ((size_t)b0 * NN + s0) * HH, rowstride);
    }
    wait_async0();
    __syncthreads();

    // hoist receiver-half A fragments (K = 128..255 of the concat) to registers
    v16bf aR[4];
#pragma unroll
    for (int k = 0; k < 4; ++k)
        aR[k] = load_frag(&sR[0][0], HH, k * 32);

    int buf = 0;
    for (int ei = e_beg; ei < e_end; ++ei) {
        // prefetch next edge's sender tile into the other buffer (overlapped)
        if (ei + 1 < e_end) {
            const int s1 = edge_index[csr_edge[ei + 1]];
            async_copy_tile_b128(&sS[buf ^ 1][0][0],
                                 hbf + ((size_t)b0 * NN + s1) * HH, rowstride);
        }

        // ---- GEMM1: T = relu(concat(h_send,h_recv) @ Wm1^T + bm1), K=256
#pragma unroll
        for (int t = 0; t < 2; ++t) {
            const int j = j0 + t * 16;
            v8f c = zerov8();
#pragma unroll
            for (int k = 0; k < 4; ++k) {
                v16bf a = load_frag(&sS[buf][0][0], HH, k * 32);
                v16bf b = load_frag(Wm1b + (size_t)j * 256, 256, k * 32);
                c = wmma_bf16(a, b, c);
            }
#pragma unroll
            for (int k = 0; k < 4; ++k) {
                v16bf b = load_frag(Wm1b + (size_t)j * 256, 256, 128 + k * 32);
                c = wmma_bf16(aR[k], b, c);
            }
            const int   col   = j + (lane & 15);
            const int   rbase = (lane >> 4) * 8;
            const float bias  = bm1[col];
#pragma unroll
            for (int v = 0; v < 8; ++v) {
                float xv = c[v] + bias;
                sT[rbase + v][col] = (__bf16)(xv > 0.f ? xv : 0.f);
            }
        }
        __syncthreads();

        // ---- GEMM2: acc += T @ Wm2^T, K=128 (running sum over edges)
#pragma unroll
        for (int t = 0; t < 2; ++t) {
            const int j = j0 + t * 16;
#pragma unroll
            for (int k0 = 0; k0 < 128; k0 += 32) {
                v16bf a = load_frag(&sT[0][0], MM, k0);
                v16bf b = load_frag(Wm2b + (size_t)j * 128, 128, k0);
                acc[t] = wmma_bf16(a, b, acc[t]);
            }
        }
        wait_async0();     // next sender tile fully in LDS (this wave's issues)
        __syncthreads();   // ... and everyone else's; sT also free to reuse
        buf ^= 1;
    }

    const int deg = e_end - e_beg;
#pragma unroll
    for (int t = 0; t < 2; ++t) {
        const int   j     = j0 + t * 16;
        const int   col   = j + (lane & 15);
        const int   rbase = (lane >> 4) * 8;
        const float bias  = (float)deg * bm2[col];
#pragma unroll
        for (int v = 0; v < 8; ++v) {
            float xv = acc[t][v] + bias;
            aggb[((size_t)(b0 + rbase + v) * NN + n) * MM + col] = (__bf16)xv;
        }
    }
}

// ---------------- GRU update ----------------
// block = 16 flat rows; 4 waves x 6 column tiles cover 384 gate columns.
__global__ __launch_bounds__(128) void gru_kernel(
        const __bf16* __restrict__ aggb,
        const __bf16* __restrict__ Wihb, const __bf16* __restrict__ Whhb,
        const float* __restrict__ bih, const float* __restrict__ bhh,
        float* __restrict__ hf, __bf16* __restrict__ hbf) {
    const int row0 = blockIdx.x * 16;
    const int wave = threadIdx.x >> 5;
    const int lane = threadIdx.x & 31;

    __shared__ float sgi[16][G3];   // 24 KB
    __shared__ float sgh[16][G3];   // 24 KB

    // hoisted A fragments (agg row tile and h row tile), K = 0..127
    v16bf aA[4], aH[4];
#pragma unroll
    for (int k = 0; k < 4; ++k) {
        aA[k] = load_frag(aggb + (size_t)row0 * MM, MM, k * 32);
        aH[k] = load_frag(hbf + (size_t)row0 * HH, HH, k * 32);
    }

#pragma unroll
    for (int t = 0; t < 6; ++t) {
        const int j = wave * 96 + t * 16;
        v8f ci = zerov8();
        v8f ch = zerov8();
#pragma unroll
        for (int k = 0; k < 4; ++k) {
            ci = wmma_bf16(aA[k], load_frag(Wihb + (size_t)j * HH, HH, k * 32), ci);
            ch = wmma_bf16(aH[k], load_frag(Whhb + (size_t)j * HH, HH, k * 32), ch);
        }
        const int   col   = j + (lane & 15);
        const int   rbase = (lane >> 4) * 8;
        const float bi = bih[col];
        const float bh = bhh[col];
#pragma unroll
        for (int v = 0; v < 8; ++v) {
            sgi[rbase + v][col] = ci[v] + bi;
            sgh[rbase + v][col] = ch[v] + bh;
        }
    }
    __syncthreads();

    // element-wise gates: thread = one hidden column, 16 rows
    const int c = threadIdx.x;  // 0..127
    for (int r = 0; r < 16; ++r) {
        float gr = sgi[r][c]       + sgh[r][c];
        float gz = sgi[r][128 + c] + sgh[r][128 + c];
        float rg = 1.f / (1.f + __expf(-gr));
        float zg = 1.f / (1.f + __expf(-gz));
        float ng = tanhf(sgi[r][256 + c] + rg * sgh[r][256 + c]);
        size_t idx = (size_t)(row0 + r) * HH + c;
        float hnew = (1.f - zg) * ng + zg * hf[idx];
        hf[idx]  = hnew;
        hbf[idx] = (__bf16)hnew;
    }
}

// ---------------- output head ----------------
__global__ __launch_bounds__(128) void out_kernel(
        const __bf16* __restrict__ hbf,
        const __bf16* __restrict__ Wo1b, const float* __restrict__ bo1,
        const float* __restrict__ Wo2, const float* __restrict__ bo2,
        float* __restrict__ out) {
    const int row0 = blockIdx.x * 16;
    const int wave = threadIdx.x >> 5;
    const int lane = threadIdx.x & 31;

    __shared__ float shid[16][HH];  // 8 KB

    v16bf aH[4];
#pragma unroll
    for (int k = 0; k < 4; ++k)
        aH[k] = load_frag(hbf + (size_t)row0 * HH, HH, k * 32);

#pragma unroll
    for (int t = 0; t < 2; ++t) {
        const int j = wave * 32 + t * 16;
        v8f c = zerov8();
#pragma unroll
        for (int k = 0; k < 4; ++k)
            c = wmma_bf16(aH[k], load_frag(Wo1b + (size_t)j * HH, HH, k * 32), c);
        const int col   = j + (lane & 15);
        const int rbase = (lane >> 4) * 8;
        const float bias = bo1[col];
#pragma unroll
        for (int v = 0; v < 8; ++v) {
            float xv = c[v] + bias;
            shid[rbase + v][col] = xv > 0.f ? xv : 0.f;
        }
    }
    __syncthreads();

    for (int idx = threadIdx.x; idx < 16 * 9; idx += 128) {
        int r = idx / 9;
        int o = idx % 9;
        float s = bo2[o];
        for (int cc = 0; cc < HH; ++cc)
            s += shid[r][cc] * Wo2[o * HH + cc];
        out[(size_t)(row0 + r) * 9 + o] = s;
    }
}

// ---------------- host launcher ----------------
extern "C" void kernel_launch(void* const* d_in, const int* in_sizes, int n_in,
                              void* d_out, int out_size, void* d_ws, size_t ws_size,
                              hipStream_t stream) {
    const float* x    = (const float*)d_in[0];
    const int*   eidx = (const int*)d_in[1];
    const float* W_in = (const float*)d_in[2];
    const float* b_in = (const float*)d_in[3];
    const float* W_m1 = (const float*)d_in[4];
    const float* b_m1 = (const float*)d_in[5];
    const float* W_m2 = (const float*)d_in[6];
    const float* b_m2 = (const float*)d_in[7];
    const float* W_ih = (const float*)d_in[8];
    const float* W_hh = (const float*)d_in[9];
    const float* b_ih = (const float*)d_in[10];
    const float* b_hh = (const float*)d_in[11];
    const float* W_o1 = (const float*)d_in[12];
    const float* b_o1 = (const float*)d_in[13];
    const float* W_o2 = (const float*)d_in[14];
    const float* b_o2 = (const float*)d_in[15];
    float* out = (float*)d_out;

    // workspace carve-up (16B-aligned chunks except csr_edge, which is 4B-aligned)
    char* w = (char*)d_ws;
    float*  hf    = (float*)w;   w += (size_t)RR * HH * 4;
    __bf16* hbf   = (__bf16*)w;  w += (size_t)RR * HH * 2;
    __bf16* aggb  = (__bf16*)w;  w += (size_t)RR * MM * 2;
    __bf16* Wm1b  = (__bf16*)w;  w += (size_t)MM * 2 * HH * 2;   // 128x256
    __bf16* Wm2b  = (__bf16*)w;  w += (size_t)MM * MM * 2;       // 128x128
    __bf16* Wihb  = (__bf16*)w;  w += (size_t)G3 * MM * 2;       // 384x128
    __bf16* Whhb  = (__bf16*)w;  w += (size_t)G3 * HH * 2;       // 384x128
    __bf16* Wo1b  = (__bf16*)w;  w += (size_t)HH * HH * 2;       // 128x128
    int* csr_ptr  = (int*)w;     w += (NN + 2) * 4;
    int* csr_edge = (int*)w;     w += EE * 4;

    build_csr_kernel<<<1, 1, 0, stream>>>(eidx, csr_ptr, csr_edge);

    cvt_bf16_kernel<<<(MM * 2 * HH + 255) / 256, 256, 0, stream>>>(W_m1, Wm1b, MM * 2 * HH);
    cvt_bf16_kernel<<<(MM * MM + 255) / 256, 256, 0, stream>>>(W_m2, Wm2b, MM * MM);
    cvt_bf16_kernel<<<(G3 * MM + 255) / 256, 256, 0, stream>>>(W_ih, Wihb, G3 * MM);
    cvt_bf16_kernel<<<(G3 * HH + 255) / 256, 256, 0, stream>>>(W_hh, Whhb, G3 * HH);
    cvt_bf16_kernel<<<(HH * HH + 255) / 256, 256, 0, stream>>>(W_o1, Wo1b, HH * HH);

    {
        size_t total = (size_t)RR * HH;
        inproj_kernel<<<(unsigned)((total + 255) / 256), 256, 0, stream>>>(x, W_in, b_in, hf, hbf);
    }

    const int rowTiles = RR / 16;  // 5184
    for (int step = 0; step < NSTEPS; ++step) {
        msg_agg_kernel<<<dim3(NN, BB / 16), 128, 0, stream>>>(
            hbf, Wm1b, Wm2b, b_m1, b_m2, eidx, csr_ptr, csr_edge, aggb);
        gru_kernel<<<rowTiles, 128, 0, stream>>>(
            aggb, Wihb, Whhb, b_ih, b_hh, hf, hbf);
    }

    out_kernel<<<rowTiles, 128, 0, stream>>>(hbf, Wo1b, b_o1, W_o2, b_o2, out);
}
